// lmrk_encoder_h_8443905704056
// MI455X (gfx1250) — matrix operand order repaired and checked by reading, not compiled, not run on hardware
//
#include <hip/hip_runtime.h>
#include <hip/hip_bf16.h>
#include <math.h>

typedef __attribute__((ext_vector_type(16))) __bf16       v16bf;
typedef __attribute__((ext_vector_type(8)))  float        v8f;
typedef __attribute__((ext_vector_type(4)))  unsigned int u32x4;

#define NB     30720
#define NODESN 68
#define DIM    128
#define EPG    204

// LDS byte offsets (all 16B aligned)
#define OFF_H     0          // float h[68][128]         34816 B
#define OFF_AGG   34816      // float agg[68][128]       34816 B
#define OFF_HCAT  69632      // bf16 hcat[80][256]       40960 B  (pool: h3T/sT/t)
#define OFF_H3T   69632      // bf16 h3T[128][96]        24576 B
#define OFF_ST    94208      // bf16 sT[16][96]          3072 B
#define OFF_T     97280      // float t[68][16]          4352 B
#define OFF_SS    110592     // float s_s[68][16]        4352 B
#define OFF_X     114944     // float xs[68][2]          544 B
#define OFF_W1    115488     // float w1s[640]           2560 B
#define OFF_EDGE  118048     // ushort eL[2*204]         816 B
#define OFF_RED   118864     // float red[256]           1024 B
#define OFF_ADJ   119888     // float adj[68][68]        18496 B (async-filled)
#define SMEM_BYTES 138384

__device__ __forceinline__ unsigned short f2bf(float f) {
  unsigned int u = __float_as_uint(f);
  u += 0x7FFFu + ((u >> 16) & 1u);            // round-to-nearest-even
  return (unsigned short)(u >> 16);
}

union Frag { v16bf v; u32x4 q[2]; };

__device__ __forceinline__ v8f wmma_bf16(v16bf a, v16bf b, v8f c) {
  return __builtin_amdgcn_wmma_f32_16x16x32_bf16(false, a, false, b, (short)0, c,
                                                 false, false);
}

// [80x256] bf16 A (LDS) @ [256x128] bf16 B (global, stored [n][k]) + bias, relu -> hout f32 [68][128]
__device__ __forceinline__ void gemm_k256_relu(const unsigned short* hcat,
                                               const unsigned short* wT,
                                               const float* bias, float* hout, int tid) {
  const int lane = tid & 31;
  const int wave = tid >> 5;          // 8 waves = 8 N-tiles
  const int half = lane >> 4;
  const int l15  = lane & 15;
  const int ncol = wave * 16 + l15;
  const float bi = bias[ncol];
  v8f acc[5];
#pragma unroll
  for (int m = 0; m < 5; ++m) acc[m] = (v8f){bi, bi, bi, bi, bi, bi, bi, bi};
#pragma unroll
  for (int ks = 0; ks < 8; ++ks) {
    Frag bf;
    const u32x4* bp = (const u32x4*)(wT + (size_t)ncol * 256 + ks * 32 + half * 16);
    bf.q[0] = bp[0];
    bf.q[1] = bp[1];
#pragma unroll
    for (int m = 0; m < 5; ++m) {
      Frag af;
      const unsigned short* ap = hcat + (m * 16 + l15) * 256 + ks * 32 + half * 16;
      af.q[0] = *(const u32x4*)ap;
      af.q[1] = *(const u32x4*)(ap + 8);
      acc[m] = wmma_bf16(af.v, bf.v, acc[m]);
    }
  }
#pragma unroll
  for (int m = 0; m < 5; ++m) {
#pragma unroll
    for (int r = 0; r < 8; ++r) {
      int row = m * 16 + half * 8 + r;
      if (row < NODESN) hout[row * DIM + ncol] = fmaxf(acc[m][r], 0.0f);
    }
  }
}

__global__ void prep_weights(const float* __restrict__ w2r, const float* __restrict__ w2l,
                             const float* __restrict__ w3r, const float* __restrict__ w3l,
                             unsigned short* __restrict__ wT2, unsigned short* __restrict__ wT3) {
  int idx = blockIdx.x * blockDim.x + threadIdx.x;   // 0..32767
  if (idx >= 128 * 256) return;
  int n = idx >> 8;
  int k = idx & 255;
  float v2 = (k < 128) ? w2r[k * 128 + n] : w2l[(k - 128) * 128 + n];
  float v3 = (k < 128) ? w3r[k * 128 + n] : w3l[(k - 128) * 128 + n];
  wT2[n * 256 + k] = f2bf(v2);
  wT3[n * 256 + k] = f2bf(v3);
}

__global__ void finalize_kernel(const float* __restrict__ ws, float* out_l1, float* out_e1) {
  if (threadIdx.x == 0) {
    out_l1[0] = sqrtf(ws[0]) * (1.0f / 142049280.0f);   // / adj.size (30720*68*68)
    out_e1[0] = ws[1] * (1.0f / 2088960.0f);            // mean over B*NODES
  }
}

__global__ __launch_bounds__(256)
void fused_graph_kernel(const float* __restrict__ x, const int* __restrict__ ei,
                        const float* __restrict__ adj, const float* __restrict__ s_in,
                        const float* __restrict__ pos,
                        const float* __restrict__ w1r, const float* __restrict__ w1l,
                        const float* __restrict__ b1,
                        const float* __restrict__ b2, const float* __restrict__ b3,
                        const unsigned short* __restrict__ wT2,
                        const unsigned short* __restrict__ wT3,
                        float* __restrict__ out_x, float* __restrict__ out_adj,
                        float* __restrict__ out_pos, float* ws_acc) {
  extern __shared__ char smem[];
  float*          h    = (float*)(smem + OFF_H);
  float*          agg  = (float*)(smem + OFF_AGG);
  unsigned short* hcat = (unsigned short*)(smem + OFF_HCAT);
  unsigned short* h3T  = (unsigned short*)(smem + OFF_H3T);
  unsigned short* sT   = (unsigned short*)(smem + OFF_ST);
  float*          t_s  = (float*)(smem + OFF_T);
  float*          s_s  = (float*)(smem + OFF_SS);
  float*          xs   = (float*)(smem + OFF_X);
  float*          w1s  = (float*)(smem + OFF_W1);
  unsigned short* eL   = (unsigned short*)(smem + OFF_EDGE);
  float*          red  = (float*)(smem + OFF_RED);
  float*          adjs = (float*)(smem + OFF_ADJ);

  const int g   = blockIdx.x;
  const int tid = threadIdx.x;
  const long Etot = (long)NB * NODESN * 3;
  const int* esrc = ei;
  const int* edst = ei + Etot;

  // ---- issue async HBM->LDS copy of this graph's adj tile NOW; it overlaps
  // the three conv layers and is retired just before the pooling phase.
  // adj tile = 68*68 f32 = 18496 B = 1156 x 16B chunks.
  {
    unsigned lds_base = (unsigned)(uintptr_t)adjs;
    const char* gbase = (const char*)adj + (size_t)g * (NODESN * NODESN * 4);
    for (int i = tid; i < 1156; i += 256) {
      unsigned           loff = lds_base + i * 16;
      unsigned long long ga   = (unsigned long long)(uintptr_t)(gbase + i * 16);
      asm volatile("global_load_async_to_lds_b128 %0, %1, off"
                   :: "v"(loff), "v"(ga) : "memory");
    }
  }

  // ---- stage-in: x, W1/b1, edges ----
  for (int i = tid; i < NODESN * 2; i += 256) xs[i] = x[(size_t)g * NODESN * 2 + i];
  for (int i = tid; i < 640; i += 256)
    w1s[i] = (i < 256) ? w1r[i] : (i < 512) ? w1l[i - 256] : b1[i - 512];
  for (int e = tid; e < EPG; e += 256) {
    eL[e]       = (unsigned short)(esrc[(size_t)g * EPG + e] - g * NODESN);
    eL[EPG + e] = (unsigned short)(edst[(size_t)g * EPG + e] - g * NODESN);
  }
  if (tid < NODESN * 2) agg[tid] = 0.0f;
  __syncthreads();

  // ---- layer 1 (K=2, scalar) ----
  for (int e = tid; e < EPG; e += 256) {
    int sl = eL[e], dl = eL[EPG + e];
    atomicAdd(&agg[dl * 2 + 0], xs[sl * 2 + 0]);
    atomicAdd(&agg[dl * 2 + 1], xs[sl * 2 + 1]);
  }
  __syncthreads();
  for (int i = tid; i < NODESN * DIM; i += 256) {
    int n = i >> 7, d = i & 127;
    float v = xs[n * 2] * w1s[d] + xs[n * 2 + 1] * w1s[128 + d] +
              agg[n * 2] * w1s[256 + d] + agg[n * 2 + 1] * w1s[384 + d] + w1s[512 + d];
    h[i] = fmaxf(v, 0.0f);
  }
  __syncthreads();

  // ---- layers 2 & 3: LDS segment-sum + bf16 WMMA GEMM [h|agg]@[Wr;Wl] ----
#pragma unroll 1
  for (int layer = 0; layer < 2; ++layer) {
    for (int i = tid; i < NODESN * DIM; i += 256) agg[i] = 0.0f;
    __syncthreads();
    for (int i = tid; i < EPG * DIM; i += 256) {
      int e = i >> 7, d = i & 127;
      atomicAdd(&agg[eL[EPG + e] * DIM + d], h[eL[e] * DIM + d]);
    }
    __syncthreads();
    for (int i = tid; i < 80 * 256; i += 256) {
      int row = i >> 8, col = i & 255;
      float v = 0.0f;
      if (row < NODESN) v = (col < DIM) ? h[row * DIM + col] : agg[row * DIM + col - DIM];
      hcat[i] = f2bf(v);
    }
    __syncthreads();
    gemm_k256_relu(hcat, layer == 0 ? wT2 : wT3, layer == 0 ? b2 : b3, h, tid);
    __syncthreads();
  }

  // ---- pooling ----
  // retire the async adj copy issued at kernel entry
  asm volatile("s_wait_asynccnt 0" ::: "memory");
  for (int i = tid; i < 16 * 96; i += 256) sT[i] = 0;
  __syncthreads();

  float ent_acc = 0.0f;
  if (tid < NODESN) {
    const float* srow = s_in + ((size_t)g * NODESN + tid) * 16;
    float v[16];
    float mx = -1e30f;
#pragma unroll
    for (int c = 0; c < 16; ++c) { v[c] = srow[c]; mx = fmaxf(mx, v[c]); }
    float sum = 0.0f;
#pragma unroll
    for (int c = 0; c < 16; ++c) { v[c] = expf(v[c] - mx); sum += v[c]; }
    float inv = 1.0f / sum;
#pragma unroll
    for (int c = 0; c < 16; ++c) {
      float p = v[c] * inv;
      s_s[tid * 16 + c] = p;
      sT[c * 96 + tid]  = f2bf(p);
      ent_acc -= p * logf(p + 1e-15f);
    }
  }
  __syncthreads();

  // t = adj @ s  (f32, small)
  for (int i = tid; i < NODESN * 16; i += 256) {
    int n = i >> 4, c = i & 15;
    float acc = 0.0f;
    for (int m = 0; m < NODESN; ++m) acc += adjs[n * NODESN + m] * s_s[m * 16 + c];
    t_s[i] = acc;
  }
  // h3 transposed to bf16 [128][96] (K=n padded to 96)
  for (int i = tid; i < 128 * 96; i += 256) {
    int d = i / 96, n = i - d * 96;
    h3T[i] = (n < NODESN) ? f2bf(h[n * DIM + d]) : (unsigned short)0;
  }
  __syncthreads();

  // out = s^T @ h3 : [16 x 128] via WMMA (M=16, K=96, N=128 over 8 waves)
  {
    const int lane = tid & 31, wave = tid >> 5;
    const int half = lane >> 4, l15 = lane & 15;
    const int ncol = wave * 16 + l15;
    v8f acc = (v8f){0, 0, 0, 0, 0, 0, 0, 0};
#pragma unroll
    for (int ks = 0; ks < 3; ++ks) {
      Frag bf, af;
      const unsigned short* bp = h3T + ncol * 96 + ks * 32 + half * 16;
      bf.q[0] = *(const u32x4*)bp;
      bf.q[1] = *(const u32x4*)(bp + 8);
      const unsigned short* ap = sT + l15 * 96 + ks * 32 + half * 16;
      af.q[0] = *(const u32x4*)ap;
      af.q[1] = *(const u32x4*)(ap + 8);
      acc = wmma_bf16(af.v, bf.v, acc);
    }
#pragma unroll
    for (int r = 0; r < 8; ++r) {
      int c = half * 8 + r;
      out_x[(size_t)g * 16 * DIM + c * DIM + ncol] = acc[r];
    }
  }

  // out_adj = s^T @ t : 16x16, one element per thread
  {
    int c = tid >> 4, k = tid & 15;
    float acc = 0.0f;
    for (int n = 0; n < NODESN; ++n) acc += s_s[n * 16 + c] * t_s[n * 16 + k];
    out_adj[(size_t)g * 256 + tid] = acc;
  }

  // link loss partial: sum (adj - s s^T)^2
  float link_acc = 0.0f;
  for (int i = tid; i < NODESN * NODESN; i += 256) {
    int n = i / NODESN, m = i - n * NODESN;
    float dot = 0.0f;
#pragma unroll
    for (int c = 0; c < 16; ++c) dot += s_s[n * 16 + c] * s_s[m * 16 + c];
    float e = adjs[i] - dot;
    link_acc += e * e;
  }

  // pos pass-through
  for (int i = tid; i < NODESN * 2; i += 256)
    out_pos[(size_t)g * NODESN * 2 + i] = pos[(size_t)g * NODESN * 2 + i];

  // block reductions -> global accumulators
  red[tid] = link_acc;
  __syncthreads();
  for (int s2 = 128; s2 > 0; s2 >>= 1) {
    if (tid < s2) red[tid] += red[tid + s2];
    __syncthreads();
  }
  if (tid == 0) atomicAdd(&ws_acc[0], red[0]);
  __syncthreads();
  red[tid] = ent_acc;
  __syncthreads();
  for (int s2 = 128; s2 > 0; s2 >>= 1) {
    if (tid < s2) red[tid] += red[tid + s2];
    __syncthreads();
  }
  if (tid == 0) atomicAdd(&ws_acc[1], red[0]);
}

extern "C" void kernel_launch(void* const* d_in, const int* in_sizes, int n_in,
                              void* d_out, int out_size, void* d_ws, size_t ws_size,
                              hipStream_t stream) {
  const float* x    = (const float*)d_in[0];
  const int*   ei   = (const int*)d_in[1];
  const float* adj  = (const float*)d_in[2];
  const float* s_in = (const float*)d_in[3];
  const float* pos  = (const float*)d_in[4];
  const float* w1r  = (const float*)d_in[5];
  const float* w1l  = (const float*)d_in[6];
  const float* b1   = (const float*)d_in[7];
  const float* w2r  = (const float*)d_in[8];
  const float* w2l  = (const float*)d_in[9];
  const float* b2   = (const float*)d_in[10];
  const float* w3r  = (const float*)d_in[11];
  const float* w3l  = (const float*)d_in[12];
  const float* b3   = (const float*)d_in[13];

  float* out     = (float*)d_out;
  float* out_x   = out;                                  // [30720,16,128]
  float* out_adj = out + (size_t)NB * 16 * DIM;          // [30720,16,16]
  float* out_l1  = out_adj + (size_t)NB * 256;           // scalar
  float* out_e1  = out_l1 + 1;                           // scalar
  float* out_pos = out_e1 + 1;                           // [N,2]

  float*          ws_acc = (float*)d_ws;
  unsigned short* wT2    = (unsigned short*)((char*)d_ws + 64);
  unsigned short* wT3    = (unsigned short*)((char*)d_ws + 64 + 128 * 256 * 2);

  hipMemsetAsync(d_ws, 0, 64, stream);
  prep_weights<<<128, 256, 0, stream>>>(w2r, w2l, w3r, w3l, wT2, wT3);
  fused_graph_kernel<<<NB, 256, SMEM_BYTES, stream>>>(
      x, ei, adj, s_in, pos, w1r, w1l, b1, b2, b3, wT2, wT3,
      out_x, out_adj, out_pos, ws_acc);
  finalize_kernel<<<1, 32, 0, stream>>>(ws_acc, out_l1, out_e1);
}
